// SpectralSelfAttention_26388279067184
// MI455X (gfx1250) — matrix operand order repaired and checked by reading, not compiled, not run on hardware
//
#include <hip/hip_runtime.h>
#include <cstdint>

typedef __attribute__((ext_vector_type(16))) _Float16 v16h;
typedef __attribute__((ext_vector_type(8)))  _Float16 v8h;
typedef __attribute__((ext_vector_type(8)))  float    v8f;
typedef __attribute__((ext_vector_type(4)))  int      v4i;

#define ROWS 32768   // B*H*W = 8*64*64
#define HW   4096
#define CCH  256
#define DQK  32
#define DV   128

static __device__ __forceinline__ v16h pack16(v8h lo, v8h hi) {
    return __builtin_shufflevector(lo, hi, 0,1,2,3,4,5,6,7,8,9,10,11,12,13,14,15);
}

static __device__ __forceinline__ v8f wmma16(v16h a, v16h b, v8f c) {
    // D = A(16x32 f16) * B(32x16 f16) + C(16x16 f32)
    return __builtin_amdgcn_wmma_f32_16x16x32_f16(false, a, false, b, (short)0, c, false, false);
}

static __device__ __forceinline__ float redmax16(float v) {
#pragma unroll
    for (int m = 1; m < 16; m <<= 1) v = fmaxf(v, __shfl_xor(v, m, 16));
    return v;
}
static __device__ __forceinline__ float redsum16(float v) {
#pragma unroll
    for (int m = 1; m < 16; m <<= 1) v += __shfl_xor(v, m, 16);
    return v;
}

// ---- CDNA5 async global->LDS copy (ASYNCcnt-tracked, bypasses VGPRs) -------
static __device__ __forceinline__ void async_ld_b128(const _Float16* gsrc, _Float16* ldst) {
#if __has_builtin(__builtin_amdgcn_global_load_async_to_lds_b128)
    void* g = (void*)gsrc;   // drop const
    __builtin_amdgcn_global_load_async_to_lds_b128(
        (__attribute__((address_space(1))) v4i*)g,
        (__attribute__((address_space(3))) v4i*)ldst, 0, 0);
#else
    uint32_t l = (uint32_t)(uintptr_t)ldst;   // flat LDS ptr low 32 bits = LDS offset
    asm volatile("global_load_async_to_lds_b128 %0, %1, off"
                 :: "v"(l), "v"((const void*)gsrc) : "memory");
#endif
}
static __device__ __forceinline__ void wait_async0() {
#if __has_builtin(__builtin_amdgcn_s_wait_asynccnt)
    __builtin_amdgcn_s_wait_asynccnt(0);
#else
    asm volatile("s_wait_asynccnt 0" ::: "memory");
#endif
}

// ---------------------------------------------------------------------------
// Kernel 1: spectral norm power iteration -> inv_sigma[4]
// sigma = || l2n(u @ W^T) @ W ||
// block 0: wf(256x32,uf) 1: wg(256x32,ug) 2: wh(256x128,uh) 3: wo(128x256,uo)
// ---------------------------------------------------------------------------
__global__ __launch_bounds__(256) void sa_spectral_sigma(
    const float* __restrict__ wf, const float* __restrict__ uf,
    const float* __restrict__ wg, const float* __restrict__ ug,
    const float* __restrict__ wh, const float* __restrict__ uh,
    const float* __restrict__ wo, const float* __restrict__ uo,
    float* __restrict__ inv_sigma)
{
    __shared__ float vbuf[256];
    __shared__ float red[256];
    const float* w; const float* u; int cin, cout;
    switch (blockIdx.x) {
        case 0:  w = wf; u = uf; cin = 256; cout = 32;  break;
        case 1:  w = wg; u = ug; cin = 256; cout = 32;  break;
        case 2:  w = wh; u = uh; cin = 256; cout = 128; break;
        default: w = wo; u = uo; cin = 128; cout = 256; break;
    }
    const int t = threadIdx.x;

    float vr = 0.f;
    if (t < cin) {
        const float* row = w + (size_t)t * cout;
        for (int j = 0; j < cout; ++j) vr += u[j] * row[j];
    }
    red[t] = vr * vr;
    __syncthreads();
    for (int s = 128; s > 0; s >>= 1) { if (t < s) red[t] += red[t + s]; __syncthreads(); }
    float inv = rsqrtf(fmaxf(red[0], 1e-12f));
    __syncthreads();
    vbuf[t] = vr * inv;
    __syncthreads();

    float tv = 0.f;
    if (t < cout) {
        for (int i = 0; i < cin; ++i) tv += vbuf[i] * w[(size_t)i * cout + t];
    }
    red[t] = tv * tv;
    __syncthreads();
    for (int s = 128; s > 0; s >>= 1) { if (t < s) red[t] += red[t + s]; __syncthreads(); }
    if (t == 0) {
        float ss = red[0];
        float sigma = ss * rsqrtf(fmaxf(ss, 1e-12f));
        inv_sigma[blockIdx.x] = 1.0f / sigma;
    }
}

// ---------------------------------------------------------------------------
// Kernel 2: f/g/h projections.  F,G row-major f16; H stored TRANSPOSED per
// batch: Ht[b][d][hw] so the flash kernel can async-copy contiguous key runs.
// ---------------------------------------------------------------------------
__global__ __launch_bounds__(256) void sa_proj_fgh(
    const float* __restrict__ x,
    const float* __restrict__ wf, const float* __restrict__ bf,
    const float* __restrict__ wg, const float* __restrict__ bg,
    const float* __restrict__ wh, const float* __restrict__ bh,
    const float* __restrict__ inv_sigma,
    _Float16* __restrict__ Fo, _Float16* __restrict__ Go, _Float16* __restrict__ Ht)
{
    extern __shared__ _Float16 smemP[];
    _Float16* WFt = smemP;                 // [32][256]  WFt[co*256+ci]
    _Float16* WGt = WFt + 32 * 256;        // [32][256]
    _Float16* WHt = WGt + 32 * 256;        // [128][256]
    _Float16* Tt  = WHt + 128 * 256;       // [128][136] transpose staging (padded)
    const int t = threadIdx.x;
    const float isf = inv_sigma[0], isg = inv_sigma[1], ish = inv_sigma[2];
    for (int i = t; i < 32 * 256; i += 256) {
        int co = i >> 8, ci = i & 255;
        WFt[i] = (_Float16)(wf[ci * 32 + co] * isf);
        WGt[i] = (_Float16)(wg[ci * 32 + co] * isg);
    }
    for (int i = t; i < 128 * 256; i += 256) {
        int co = i >> 8, ci = i & 255;
        WHt[i] = (_Float16)(wh[ci * 128 + co] * ish);
    }
    __syncthreads();

    const int lane = t & 31, wave = t >> 5, ln = lane & 15, grp = lane >> 4;
    const int row0 = blockIdx.x * 128 + wave * 16;

    v8f accF[2] = {}, accG[2] = {}, accH[8] = {};
    const float* xr = x + (size_t)(row0 + ln) * CCH;

#pragma unroll
    for (int kb = 0; kb < 256; kb += 32) {
        v16h a;
#pragma unroll
        for (int e = 0; e < 8; ++e) {
            a[e]     = (_Float16)xr[kb + grp * 16 + e];
            a[8 + e] = (_Float16)xr[kb + grp * 16 + 8 + e];
        }
        // F/G fragments up front (4 independent dests)
        v16h bF0 = *(const v16h*)(WFt + (ln)      * 256 + kb + grp * 16);
        v16h bF1 = *(const v16h*)(WFt + (16 + ln) * 256 + kb + grp * 16);
        v16h bG0 = *(const v16h*)(WGt + (ln)      * 256 + kb + grp * 16);
        v16h bG1 = *(const v16h*)(WGt + (16 + ln) * 256 + kb + grp * 16);
        accF[0] = wmma16(a, bF0, accF[0]);
        accF[1] = wmma16(a, bF1, accF[1]);
        accG[0] = wmma16(a, bG0, accG[0]);
        accG[1] = wmma16(a, bG1, accG[1]);
        // H: double-buffered B fragments
        v16h bh0 = *(const v16h*)(WHt + (ln) * 256 + kb + grp * 16);
#pragma unroll
        for (int nt = 0; nt < 8; ++nt) {
            v16h cur = bh0;
            if (nt < 7)
                bh0 = *(const v16h*)(WHt + ((nt + 1) * 16 + ln) * 256 + kb + grp * 16);
            accH[nt] = wmma16(a, cur, accH[nt]);
        }
    }

#pragma unroll
    for (int nt = 0; nt < 2; ++nt) {
        float bfv = bf[nt * 16 + ln], bgv = bg[nt * 16 + ln];
#pragma unroll
        for (int e = 0; e < 8; ++e) {
            int r = row0 + e + 8 * grp;
            Fo[(size_t)r * DQK + nt * 16 + ln] = (_Float16)(accF[nt][e] + bfv);
            Go[(size_t)r * DQK + nt * 16 + ln] = (_Float16)(accG[nt][e] + bgv);
        }
    }
    // H -> LDS transpose tile Tt[d][local_row], then coalesced global stores
#pragma unroll
    for (int nt = 0; nt < 8; ++nt) {
        float bhv = bh[nt * 16 + ln];
#pragma unroll
        for (int e = 0; e < 8; ++e) {
            int rl = wave * 16 + e + 8 * grp;          // 0..127
            Tt[(nt * 16 + ln) * 136 + rl] = (_Float16)(accH[nt][e] + bhv);
        }
    }
    __syncthreads();
    const int bb  = (blockIdx.x * 128) >> 12;          // batch
    const int r0g = (blockIdx.x * 128) & 4095;         // row within batch
#pragma unroll
    for (int j = 0; j < 8; ++j) {
        int c = t + 256 * j;                            // 0..2047 chunks of 8 halves
        int d = c >> 4, rr = (c & 15) * 8;
        *(v8h*)(Ht + ((size_t)bb * DV + d) * HW + r0g + rr) =
            *(const v8h*)(Tt + d * 136 + rr);
    }
}

// ---------------------------------------------------------------------------
// Kernel 3: flash attention, double-buffered async pipeline.
// grid(32 qblocks, 8 batch) x 256 thr.  Tiles stream into ping-pong LDS
// buffers via GLOBAL_LOAD_ASYNC_TO_LDS_B128 while WMMAs chew the other one.
// LDS: Fs[2][64][40] + Hts[2][128][72] + Ps[8][16][72] = 64 KB (dynamic).
// ---------------------------------------------------------------------------
#define FS_STRIDE  (64 * 40)     // halves per F buffer
#define HTS_STRIDE (128 * 72)    // halves per H buffer
#define PS_BASE    (2 * FS_STRIDE + 2 * HTS_STRIDE)

__global__ __launch_bounds__(256) void sa_flash_attn(
    const _Float16* __restrict__ Fi, const _Float16* __restrict__ Gi,
    const _Float16* __restrict__ Ht, _Float16* __restrict__ Oo)
{
    extern __shared__ _Float16 smemA[];
    _Float16* FsB  = smemA;                       // [2][64][40]
    _Float16* HtsB = smemA + 2 * FS_STRIDE;       // [2][128][72]

    const int t = threadIdx.x, lane = t & 31, wave = t >> 5, ln = lane & 15, grp = lane >> 4;
    const int bb = blockIdx.y;
    const int q0 = blockIdx.x * 128 + wave * 16;
    const size_t base = (size_t)bb * HW;
    _Float16* PsW = smemA + PS_BASE + wave * (16 * 72);   // this wave's P [16][72]

    // Q fragment (A layout), loaded once
    const _Float16* gq = Gi + (base + q0 + ln) * DQK;
    v16h qa = pack16(*(const v8h*)(gq + grp * 16), *(const v8h*)(gq + grp * 16 + 8));

    // per-thread async-copy mappings
    const int fkey = t >> 2, fc0 = (t & 3) * 8;           // F: 256 x 16B chunks
    const _Float16* fsrc = Fi + (base + fkey) * DQK + fc0;
    const size_t hrow = (size_t)bb * DV;                  // H: 1024 x 16B chunks

    v8f Oacc[8] = {};
    float rmax[8], rsum[8];
#pragma unroll
    for (int e = 0; e < 8; ++e) { rmax[e] = -1e30f; rsum[e] = 0.f; }

    // ---- prologue: issue tile 0 into buffer 0 ----
    async_ld_b128(fsrc, FsB + fkey * 40 + fc0);
#pragma unroll
    for (int j = 0; j < 4; ++j) {
        int c = t + 256 * j;
        int d = c >> 3, k0 = (c & 7) * 8;
        async_ld_b128(Ht + (hrow + d) * HW + k0, HtsB + d * 72 + k0);
    }

    for (int it = 0; it < HW / 64; ++it) {
        const int kb0 = it * 64;
        const int cur = it & 1;
        wait_async0();                                // tile `it` landed (own ops)
        __syncthreads();                              // everyone's landed; other buf free

        // ---- issue tile it+1 into the other buffer; overlaps with compute ----
        if (it + 1 < HW / 64) {
            const int nb = cur ^ 1, nk = kb0 + 64;
            async_ld_b128(fsrc + (size_t)nk * DQK, FsB + nb * FS_STRIDE + fkey * 40 + fc0);
#pragma unroll
            for (int j = 0; j < 4; ++j) {
                int c = t + 256 * j;
                int d = c >> 3, k0 = (c & 7) * 8;
                async_ld_b128(Ht + (hrow + d) * HW + nk + k0,
                              HtsB + nb * HTS_STRIDE + d * 72 + k0);
            }
            if (it + 2 < HW / 64)
                __builtin_prefetch(fsrc + (size_t)(kb0 + 128) * DQK, 0, 1);
        }

        const _Float16* Fcur = FsB + cur * FS_STRIDE;
        const _Float16* Hcur = HtsB + cur * HTS_STRIDE;

        // ---- S = Q * F^T : 4 WMMAs (K=32 = d_qk) ----
        const _Float16* f0 = Fcur + (ln)      * 40 + grp * 16;
        const _Float16* f1 = Fcur + (16 + ln) * 40 + grp * 16;
        const _Float16* f2 = Fcur + (32 + ln) * 40 + grp * 16;
        const _Float16* f3 = Fcur + (48 + ln) * 40 + grp * 16;
        v16h b0 = pack16(*(const v8h*)f0, *(const v8h*)(f0 + 8));
        v16h b1 = pack16(*(const v8h*)f1, *(const v8h*)(f1 + 8));
        v16h b2 = pack16(*(const v8h*)f2, *(const v8h*)(f2 + 8));
        v16h b3 = pack16(*(const v8h*)f3, *(const v8h*)(f3 + 8));
        v8f z = {};
        v8f s[4];
        s[0] = wmma16(qa, b0, z);
        s[1] = wmma16(qa, b1, z);
        s[2] = wmma16(qa, b2, z);
        s[3] = wmma16(qa, b3, z);

        // ---- online softmax over this 64-key block ----
        float sc[8];
#pragma unroll
        for (int e = 0; e < 8; ++e) {
            float lm = fmaxf(fmaxf(s[0][e], s[1][e]), fmaxf(s[2][e], s[3][e]));
            lm = redmax16(lm);
            float nm = fmaxf(rmax[e], lm);
            sc[e] = __expf(rmax[e] - nm);
            rmax[e] = nm;
            float ps = 0.f;
#pragma unroll
            for (int kt = 0; kt < 4; ++kt) {
                float pv = __expf(s[kt][e] - nm);
                ps += pv;
                PsW[(e + 8 * grp) * 72 + kt * 16 + ln] = (_Float16)pv;  // D -> row-major
            }
            rsum[e] = rsum[e] * sc[e] + redsum16(ps);
        }
#pragma unroll
        for (int nt = 0; nt < 8; ++nt)
#pragma unroll
            for (int e = 0; e < 8; ++e) Oacc[nt][e] *= sc[e];
        __syncthreads();                              // P visible (wave-local + tidy)

        // ---- O += P(16x64) * H(64x128): double-buffered H fragments ----
#pragma unroll
        for (int kc = 0; kc < 64; kc += 32) {
            const _Float16* pr = PsW + (ln) * 72 + kc + grp * 16;   // A layout for P
            v16h pa = pack16(*(const v8h*)pr, *(const v8h*)(pr + 8));
            const _Float16* h0 = Hcur + (ln) * 72 + kc + grp * 16;
            v16h hb = pack16(*(const v8h*)h0, *(const v8h*)(h0 + 8));
#pragma unroll
            for (int nt = 0; nt < 8; ++nt) {
                v16h curb = hb;
                if (nt < 7) {
                    const _Float16* hn = Hcur + ((nt + 1) * 16 + ln) * 72 + kc + grp * 16;
                    hb = pack16(*(const v8h*)hn, *(const v8h*)(hn + 8));
                }
                Oacc[nt] = wmma16(pa, curb, Oacc[nt]);
            }
        }
    }

#pragma unroll
    for (int nt = 0; nt < 8; ++nt)
#pragma unroll
        for (int e = 0; e < 8; ++e) {
            int r = q0 + e + 8 * grp;
            Oo[(base + r) * DV + nt * 16 + ln] = (_Float16)(Oacc[nt][e] / rsum[e]);
        }
}

// ---------------------------------------------------------------------------
// Kernel 4: out = gamma * (O @ wo_n + bo) + x.  256 blocks x 256 thr.
// ---------------------------------------------------------------------------
__global__ __launch_bounds__(256) void sa_out_proj(
    const _Float16* __restrict__ Oi,
    const float* __restrict__ wo, const float* __restrict__ bo,
    const float* __restrict__ inv_sigma, const float* __restrict__ gamma,
    const float* __restrict__ x, float* __restrict__ out)
{
    extern __shared__ _Float16 smemO[];
    _Float16* WOt = smemO;                 // [256][128]  WOt[co*128+ci]
    const int t = threadIdx.x;
    const float iso = inv_sigma[3];
    for (int i = t; i < 256 * 128; i += 256) {
        int co = i >> 7, ci = i & 127;
        WOt[i] = (_Float16)(wo[ci * 256 + co] * iso);
    }
    __syncthreads();

    const int lane = t & 31, wave = t >> 5, ln = lane & 15, grp = lane >> 4;
    const int row0 = blockIdx.x * 128 + wave * 16;

    v8f acc[16] = {};
    const _Float16* orow = Oi + (size_t)(row0 + ln) * DV;
#pragma unroll
    for (int kb = 0; kb < 128; kb += 32) {
        v16h a = pack16(*(const v8h*)(orow + kb + grp * 16),
                        *(const v8h*)(orow + kb + grp * 16 + 8));
        v16h bcur = *(const v16h*)(WOt + (ln) * 128 + kb + grp * 16);
#pragma unroll
        for (int nt = 0; nt < 16; ++nt) {
            v16h cur = bcur;
            if (nt < 15)
                bcur = *(const v16h*)(WOt + ((nt + 1) * 16 + ln) * 128 + kb + grp * 16);
            acc[nt] = wmma16(a, cur, acc[nt]);
        }
    }
    const float gm = gamma[0];
#pragma unroll
    for (int nt = 0; nt < 16; ++nt) {
        float bv = bo[nt * 16 + ln];
#pragma unroll
        for (int e = 0; e < 8; ++e) {
            int r = row0 + e + 8 * grp;
            size_t idx = (size_t)r * CCH + nt * 16 + ln;
            out[idx] = gm * (acc[nt][e] + bv) + x[idx];
        }
    }
}

// ---------------------------------------------------------------------------
extern "C" void kernel_launch(void* const* d_in, const int* in_sizes, int n_in,
                              void* d_out, int out_size, void* d_ws, size_t ws_size,
                              hipStream_t stream) {
    const float* x     = (const float*)d_in[0];
    const float* wf    = (const float*)d_in[1];
    const float* bf    = (const float*)d_in[2];
    const float* uf    = (const float*)d_in[3];
    const float* wg    = (const float*)d_in[4];
    const float* bg    = (const float*)d_in[5];
    const float* ug    = (const float*)d_in[6];
    const float* wh    = (const float*)d_in[7];
    const float* bh    = (const float*)d_in[8];
    const float* uh    = (const float*)d_in[9];
    const float* wo    = (const float*)d_in[10];
    const float* bo    = (const float*)d_in[11];
    const float* uo    = (const float*)d_in[12];
    const float* gamma = (const float*)d_in[13];

    float* inv_sigma = (float*)d_ws;
    char* p = (char*)d_ws + 256;
    _Float16* Fb  = (_Float16*)p; p += (size_t)ROWS * DQK * 2;   // 2 MB
    _Float16* Gb  = (_Float16*)p; p += (size_t)ROWS * DQK * 2;   // 2 MB
    _Float16* Htb = (_Float16*)p; p += (size_t)ROWS * DV  * 2;   // 8 MB (transposed)
    _Float16* Ob  = (_Float16*)p; p += (size_t)ROWS * DV  * 2;   // 8 MB

    sa_spectral_sigma<<<4, 256, 0, stream>>>(wf, uf, wg, ug, wh, uh, wo, uo, inv_sigma);

    // weights (96KB) + transpose tile (34KB) dynamic LDS
    sa_proj_fgh<<<ROWS / 128, 256, (192 * 256 + 128 * 136) * sizeof(_Float16), stream>>>(
        x, wf, bf, wg, bg, wh, bh, inv_sigma, Fb, Gb, Htb);

    dim3 ag(HW / 128, 8, 1);
    size_t flash_lds = (size_t)(2 * FS_STRIDE + 2 * HTS_STRIDE + 8 * 16 * 72) * sizeof(_Float16);
    sa_flash_attn<<<ag, 256, flash_lds, stream>>>(Fb, Gb, Htb, Ob);

    sa_out_proj<<<ROWS / 128, 256, 256 * 128 * sizeof(_Float16), stream>>>(
        Ob, wo, bo, inv_sigma, gamma, x, (float*)d_out);
}
